// GenealogyGNN_63067299774638
// MI455X (gfx1250) — matrix operand order repaired and checked by reading, not compiled, not run on hardware
//
#include <hip/hip_runtime.h>
#include <hip/hip_bf16.h>

typedef __attribute__((ext_vector_type(16))) _Float16 v16h;
typedef __attribute__((ext_vector_type(8)))  float    v8f;
typedef __attribute__((ext_vector_type(2)))  _Float16 h2;

#define NNODES 100000
#define NEDGES 300000
#define ETOT   (NEDGES + NNODES)
#define DH     64
#define HEADS  4
#define NEG_SLOPE 0.2f

// ---------------------------------------------------------------------------
// float atomic max via sign-split integer atomics (lowers to
// global_atomic_max_i32 / global_atomic_min_u32 — no CAS loop).
// Location must be initialized to -inf.
// ---------------------------------------------------------------------------
__device__ __forceinline__ void atomicMaxF(float* addr, float val) {
  if (val >= 0.0f)
    atomicMax((int*)addr, __float_as_int(val));
  else
    atomicMin((unsigned int*)addr, __float_as_uint(val));
}

// ---------------------------------------------------------------------------
// CDNA5 packed f16 atomic add (GLOBAL_ATOMIC_PK_ADD_F16, no return ->
// STOREcnt). Two f16 elements per atomic: halves the atomic count and the
// scatter bytes vs per-float global_atomic_add_f32.
// ---------------------------------------------------------------------------
__device__ __forceinline__ void atomicPkAddF16(h2* ptr, h2 val) {
  unsigned long long a = (unsigned long long)ptr;
  unsigned int v = __builtin_bit_cast(unsigned int, val);
  asm volatile("global_atomic_pk_add_f16 %0, %1, off"
               :: "v"(a), "v"(v)
               : "memory");
}

// ---------------------------------------------------------------------------
// WMMA GEMM:  out[r, c] = act( sum_k A[r,k] * W[c,k] + bias[c] )
// A: [nrows, 64] f32 (row-major), W: [ncols, 64] f32 (row-major, B = W^T)
// block = 128 threads = 4 waves; wave -> 16 rows x 64 cols (4 n-tiles)
// grid.x = ceil(nrows/64), grid.y = ncols/64.  nrows % 16 == 0 assumed.
// If Out16 != null, results are stored as f16 (for the edge-phase gather).
// ---------------------------------------------------------------------------
__global__ __launch_bounds__(128) void k_lin(const float* __restrict__ A,
                                             const float* __restrict__ W,
                                             const float* __restrict__ bias,
                                             float* __restrict__ Out,
                                             _Float16* __restrict__ Out16,
                                             int nrows, int ncols, int do_relu) {
  const int wave = threadIdx.x >> 5;
  const int lane = threadIdx.x & 31;
  const int half = lane >> 4;      // 0: lanes 0-15, 1: lanes 16-31
  const int m    = lane & 15;

  const int r0 = blockIdx.x * 64 + wave * 16;
  if (r0 >= nrows) return;                       // wave-uniform: EXEC stays full
  const int c0 = blockIdx.y * 64;
  const int row = r0 + m;                        // A-fragment row for this lane

  v8f acc0 = {}, acc1 = {}, acc2 = {}, acc3 = {};

#pragma unroll
  for (int kk = 0; kk < 64; kk += 32) {
    // ---- A fragment: 16x32 f16.  ISA layout:
    //   VGPR j(0..3): K = 8*half + 2j,2j+1 ; VGPR j(4..7): K = 16 + 8*half + 2(j-4)
    v16h a;
#pragma unroll
    for (int j = 0; j < 4; ++j) {
      int kb = kk + 8 * half + 2 * j;
      float2 v0 = *(const float2*)(A + (size_t)row * 64 + kb);
      a[2 * j]     = (_Float16)v0.x;
      a[2 * j + 1] = (_Float16)v0.y;
      float2 v1 = *(const float2*)(A + (size_t)row * 64 + kb + 16);
      a[8 + 2 * j] = (_Float16)v1.x;
      a[9 + 2 * j] = (_Float16)v1.y;
    }

    // ---- B fragments: 32x16 f16 per n-tile.  ISA layout:
    //   VGPR j(0..3): K = 8*half + 2j,2j+1 ; VGPR j(4..7): K = 16 + 8*half + 2(j-4) ; N = lane & 15
#pragma unroll
    for (int t = 0; t < 4; ++t) {
      const int col = c0 + 16 * t + m;
      const float* wr = W + (size_t)col * 64;
      v16h b;
#pragma unroll
      for (int j = 0; j < 8; ++j) {
        int kb = kk + 8 * half + 2 * (j & 3) + ((j >> 2) << 4);
        float2 v = *(const float2*)(wr + kb);
        b[2 * j]     = (_Float16)v.x;
        b[2 * j + 1] = (_Float16)v.y;
      }
      v8f* accp = (t == 0) ? &acc0 : (t == 1) ? &acc1 : (t == 2) ? &acc2 : &acc3;
      *accp = __builtin_amdgcn_wmma_f32_16x16x32_f16(
          /*neg_a=*/false, a, /*neg_b=*/false, b,
          /*c_mod=*/(short)0, *accp, /*reuse_a=*/false, /*reuse_b=*/false);
    }
  }

  // ---- store: C/D layout: VGPR r -> M = r + 8*half ; N = lane & 15
#pragma unroll
  for (int t = 0; t < 4; ++t) {
    const v8f accv = (t == 0) ? acc0 : (t == 1) ? acc1 : (t == 2) ? acc2 : acc3;
    const int col = c0 + 16 * t + m;
    const float bv = bias ? bias[col] : 0.0f;
#pragma unroll
    for (int r = 0; r < 8; ++r) {
      const int orow = r0 + 8 * half + r;
      float v = accv[r] + bv;
      if (do_relu) v = fmaxf(v, 0.0f);
      if (Out16) Out16[(size_t)orow * ncols + col] = (_Float16)v;
      else       Out[(size_t)orow * ncols + col] = v;
    }
  }
}

// ---------------------------------------------------------------------------
// Per (node, head): attention dot products from f16 xh; init m=-inf, z=0,
// agg16=0.
// ---------------------------------------------------------------------------
__global__ void k_alpha(const _Float16* __restrict__ xh,
                        const float* __restrict__ aw_src,
                        const float* __restrict__ aw_dst,
                        float* __restrict__ as_o, float* __restrict__ ad_o,
                        float* __restrict__ mbuf, float* __restrict__ zbuf,
                        _Float16* __restrict__ agg, int n) {
  const int i = blockIdx.x * blockDim.x + threadIdx.x;
  if (i >= n * HEADS) return;
  const int node = i >> 2, h = i & 3;
  const _Float16* xr = xh + (size_t)node * 256 + h * 64;
  const float* ws = aw_src + h * 64;
  const float* wd = aw_dst + h * 64;
  float s = 0.0f, d = 0.0f;
#pragma unroll 8
  for (int k = 0; k < 64; ++k) {
    const float xv = (float)xr[k];
    s += xv * ws[k];
    d += xv * wd[k];
  }
  as_o[i] = s;
  ad_o[i] = d;
  mbuf[i] = -__builtin_inff();
  zbuf[i] = 0.0f;
  float4* ag = (float4*)(agg + (size_t)node * 256 + h * 64);   // 128B of f16
#pragma unroll
  for (int k = 0; k < 8; ++k) ag[k] = make_float4(0.f, 0.f, 0.f, 0.f);
}

// ---------------------------------------------------------------------------
// Pass 1: e = leaky_relu(as[src]+ad[dst]); stash e; segment max by dst.
// Edge id e < NEDGES -> edge_index, else self-loop (e - NEDGES).
// ---------------------------------------------------------------------------
__global__ void k_edge_max(const int* __restrict__ ei,
                           const float* __restrict__ as_,
                           const float* __restrict__ ad_,
                           float* __restrict__ ebuf,
                           float* __restrict__ mbuf) {
  const int i = blockIdx.x * blockDim.x + threadIdx.x;
  if (i >= ETOT * HEADS) return;
  const int e = i >> 2, h = i & 3;
  int s, d;
  if (e < NEDGES) { s = ei[e]; d = ei[NEDGES + e]; }
  else            { s = d = e - NEDGES; }
  float v = as_[s * HEADS + h] + ad_[d * HEADS + h];
  v = (v > 0.0f) ? v : NEG_SLOPE * v;
  ebuf[i] = v;
  atomicMaxF(mbuf + d * HEADS + h, v);
}

// ---------------------------------------------------------------------------
// Pass 2: w = exp(e - m[dst]); overwrite ebuf with w; segment sum -> z.
// ---------------------------------------------------------------------------
__global__ void k_edge_sum(const int* __restrict__ ei,
                           const float* __restrict__ mbuf,
                           float* __restrict__ ebuf,
                           float* __restrict__ zbuf) {
  const int i = blockIdx.x * blockDim.x + threadIdx.x;
  if (i >= ETOT * HEADS) return;
  const int e = i >> 2, h = i & 3;
  int d;
  if (e < NEDGES) d = ei[NEDGES + e];
  else            d = e - NEDGES;
  const float w = __expf(ebuf[i] - mbuf[d * HEADS + h]);
  ebuf[i] = w;
  atomicAdd(zbuf + d * HEADS + h, w);
}

// ---------------------------------------------------------------------------
// Pass 3: one wave per edge; agg[dst] += xh[src] * (w / z[dst]) per head.
// f16 gather (512B/edge) + packed-f16 atomic scatter (32 atomics/head-pair).
// Each lane owns one f16x2 pair per head: element pair [h*64 + 2*lane, +1].
// ---------------------------------------------------------------------------
__global__ void k_aggr(const int* __restrict__ ei,
                       const float* __restrict__ ebuf,
                       const float* __restrict__ zbuf,
                       const _Float16* __restrict__ xh,
                       _Float16* __restrict__ agg) {
  const int gid = blockIdx.x * blockDim.x + threadIdx.x;
  const int e = gid >> 5;
  const int lane = gid & 31;
  if (e >= ETOT) return;
  int s, d;
  if (e < NEDGES) { s = ei[e]; d = ei[NEDGES + e]; }
  else            { s = d = e - NEDGES; }
  const h2* xs = (const h2*)(xh + (size_t)s * 256);
  h2* ag = (h2*)(agg + (size_t)d * 256);
#pragma unroll
  for (int h = 0; h < HEADS; ++h) {
    const float coef = ebuf[e * HEADS + h] / (zbuf[d * HEADS + h] + 1e-16f);
    const h2 xv = xs[h * 32 + lane];
    h2 r;
    r.x = (_Float16)((float)xv.x * coef);
    r.y = (_Float16)((float)xv.y * coef);
    atomicPkAddF16(ag + h * 32 + lane, r);
  }
}

// ---------------------------------------------------------------------------
// Finalize: out = relu(mean_heads(agg) + bias) [+ residual if layer > 0]
// ---------------------------------------------------------------------------
__global__ void k_final(const _Float16* __restrict__ agg,
                        const float* __restrict__ bias,
                        const float* __restrict__ hres,
                        float* __restrict__ out, int n, int use_res) {
  const int i = blockIdx.x * blockDim.x + threadIdx.x;
  if (i >= n * DH) return;
  const int node = i >> 6, dd = i & 63;
  const _Float16* ag = agg + (size_t)node * 256;
  float v = 0.25f * ((float)ag[dd] + (float)ag[64 + dd] +
                     (float)ag[128 + dd] + (float)ag[192 + dd]) + bias[dd];
  v = fmaxf(v, 0.0f);
  if (use_res) v += hres[i];
  out[i] = v;
}

// ---------------------------------------------------------------------------
// Host orchestration
// ---------------------------------------------------------------------------
extern "C" void kernel_launch(void* const* d_in, const int* in_sizes, int n_in,
                              void* d_out, int out_size, void* d_ws, size_t ws_size,
                              hipStream_t stream) {
  const float* x       = (const float*)d_in[0];  // [N,64]
  const int*   ei      = (const int*)  d_in[1];  // [2,E]
  // d_in[2] = edge_type (unused by the reference computation)
  const float* W_in    = (const float*)d_in[3];  // [64,64]
  const float* b_in    = (const float*)d_in[4];  // [64]
  const float* W_g     = (const float*)d_in[5];  // [3,256,64]
  const float* att_src = (const float*)d_in[6];  // [3,4,64]
  const float* att_dst = (const float*)d_in[7];  // [3,4,64]
  const float* b_g     = (const float*)d_in[8];  // [3,64]
  float* out = (float*)d_out;                    // [N,64]

  const int N = NNODES;

  // workspace carve-up
  char* wsb = (char*)d_ws;
  size_t off = 0;
  float* hA = (float*)(wsb + off);      off += (size_t)N * 64 * sizeof(float);
  float* hB = (float*)(wsb + off);      off += (size_t)N * 64 * sizeof(float);
  _Float16* xh16 = (_Float16*)(wsb + off); off += (size_t)N * 256 * sizeof(_Float16);
  _Float16* agg16 = (_Float16*)(wsb + off); off += (size_t)N * 256 * sizeof(_Float16);
  float* asb  = (float*)(wsb + off);    off += (size_t)N * HEADS * sizeof(float);
  float* adb  = (float*)(wsb + off);    off += (size_t)N * HEADS * sizeof(float);
  float* mb   = (float*)(wsb + off);    off += (size_t)N * HEADS * sizeof(float);
  float* zb   = (float*)(wsb + off);    off += (size_t)N * HEADS * sizeof(float);
  float* ebuf = (float*)(wsb + off);    off += (size_t)ETOT * HEADS * sizeof(float);
  (void)ws_size; (void)in_sizes; (void)n_in; (void)out_size;

  const int gemm_rb = (N + 63) / 64;          // 1563 row-blocks
  const int nh_blk  = (N * HEADS + 255) / 256;
  const int eh_blk  = (ETOT * HEADS + 255) / 256;
  const int agg_blk = (ETOT * 32 + 255) / 256;
  const int fin_blk = (N * DH + 255) / 256;

  // input projection: h0 = relu(x @ W_in^T + b_in)  (f32 out)
  k_lin<<<dim3(gemm_rb, 1), 128, 0, stream>>>(x, W_in, b_in, hA, nullptr, N, 64, 1);

  for (int layer = 0; layer < 3; ++layer) {
    const float* h_in = (layer == 1) ? hB : hA;
    float* h_out = (layer == 0) ? hB : (layer == 1) ? hA : out;

    // xh = h_in @ W_g[layer]^T  -> [N, 256] stored f16 for the edge phase
    k_lin<<<dim3(gemm_rb, 4), 128, 0, stream>>>(
        h_in, W_g + (size_t)layer * 256 * 64, nullptr, nullptr, xh16, N, 256, 0);

    // attention scalars + buffer init
    k_alpha<<<nh_blk, 256, 0, stream>>>(
        xh16, att_src + layer * HEADS * 64, att_dst + layer * HEADS * 64,
        asb, adb, mb, zb, agg16, N);

    // segment softmax over incoming edges
    k_edge_max<<<eh_blk, 256, 0, stream>>>(ei, asb, adb, ebuf, mb);
    k_edge_sum<<<eh_blk, 256, 0, stream>>>(ei, mb, ebuf, zb);

    // weighted aggregation (f16 gather + packed f16 atomics)
    k_aggr<<<agg_blk, 256, 0, stream>>>(ei, ebuf, zb, xh16, agg16);

    // head mean + bias + relu (+ residual after layer 0)
    k_final<<<fin_blk, 256, 0, stream>>>(
        agg16, b_g + layer * 64, h_in, h_out, N, layer > 0 ? 1 : 0);
  }
}
